// NodePropLayer_46643344835303
// MI455X (gfx1250) — matrix operand order repaired and checked by reading, not compiled, not run on hardware
//
#include <hip/hip_runtime.h>
#include <hip/hip_bf16.h>

typedef __attribute__((ext_vector_type(16))) __bf16 v16bf;
typedef __attribute__((ext_vector_type(8)))  float  v8f;
typedef __attribute__((ext_vector_type(4)))  unsigned int u32x4;
typedef __attribute__((ext_vector_type(8)))  int i32x8;
typedef __attribute__((ext_vector_type(4)))  int i32x4;

#define NODE_D 128
// fragment sizes (elements) of pre-swizzled bf16 weights in d_ws
#define S_EDGE (12 * 8 * 512)   // 49152: K=384 -> 12 ksteps, 8 ntiles, 512 bf16/frag
#define S_UPD  (4  * 8 * 512)   // 16384: K=128 -> 4 ksteps

#if defined(__has_builtin)
#if __has_builtin(__builtin_amdgcn_tensor_load_to_lds)
#define HAVE_TDM 1
#endif
#endif
#if __has_include(<hip/amd_detail/amd_gfx1250_TDM.h>)
#define TDM_6ARG 1
#endif

// ---------------------------------------------------------------------------
// Pre-swizzle weights into per-lane WMMA B-fragment layout (bf16).
// B 32x16 bf16 layout (ISA 7.12.2): lane l -> column n = l%16 ; K-half by l/16;
// within a lane, element j -> k_local = (j/8)*16 + (l>=16)*8 + j%8.
// Fragment (ks, nt) stored as 512 bf16, lane-major (16 contiguous per lane).
// ---------------------------------------------------------------------------
__global__ __launch_bounds__(256) void prep_weights(
    const float* __restrict__ Wmsg, const float* __restrict__ Wrev,
    const float* __restrict__ Wupd, __bf16* __restrict__ ws)
{
    int idx = blockIdx.x * blockDim.x + threadIdx.x;
    const float* W;
    __bf16* dst;
    int local;
    if (idx < S_EDGE)               { W = Wmsg; dst = ws;            local = idx; }
    else if (idx < 2 * S_EDGE)      { W = Wrev; dst = ws + S_EDGE;   local = idx - S_EDGE; }
    else if (idx < 2*S_EDGE + S_UPD){ W = Wupd; dst = ws + 2*S_EDGE; local = idx - 2*S_EDGE; }
    else return;

    int ks   = local >> 12;          // / 4096  (8 ntiles * 512)
    int rem  = local & 4095;
    int nt   = rem >> 9;             // / 512
    int q    = rem & 511;
    int lane = q >> 4;
    int j    = q & 15;
    int k = ks * 32 + ((j >> 3) << 4) + ((lane >> 4) << 3) + (j & 7);   // kstep*32 + (j/8)*16 + khalf*8 + j%8
    int n = nt * 16 + (lane & 15);
    dst[local] = (__bf16)W[k * NODE_D + n];     // W is [K,128] row-major
}

// ---------------------------------------------------------------------------
// TDM: DMA a contiguous block (nbytes, 8B-aligned) from global to LDS.
// D# per CDNA5 ISA: 1-D copy as a 2-D tile,
// data_size=3 (8B elems), tile_dim0 = tensor_dim0 = nbytes/8 (<= 65535),
// tile_dim1 = tensor_dim1 = 1, groups 2/3 zero (<=2D), type=2.
// ---------------------------------------------------------------------------
#if defined(HAVE_TDM)
__device__ __forceinline__ void tdm_load_1d(unsigned lds_addr, const void* gptr,
                                            unsigned nbytes)
{
    unsigned long long ga = (unsigned long long)(uintptr_t)gptr;
    unsigned n8 = nbytes >> 3;                       // 8-byte elements
    u32x4 g0;
    g0[0] = 1u;                                      // count=1 (valid), user mode
    g0[1] = lds_addr;                                // lds_addr -> bits[63:32]
    g0[2] = (unsigned)ga;                            // global_addr[31:0]
    g0[3] = (unsigned)((ga >> 32) & 0x01ffffffu)     // global_addr[56:32]
          | (2u << 30);                              // type=2 ("image")
    i32x8 g1;
    g1[0] = (int)(3u << 16);                         // workgroup_mask=0, data_size=3 (8B)
    g1[1] = (int)((n8 & 0xffffu) << 16);             // tensor_dim0[15:0] @ bits[63:48]
    g1[2] = (int)(1u << 16);                         // tensor_dim0 hi=0, tensor_dim1=1
    g1[3] = (int)((n8 & 0xffffu) << 16);             // tensor_dim1 hi=0, tile_dim0=n8
    g1[4] = 1;                                       // tile_dim1=1, tile_dim2=0
    g1[5] = (int)n8;                                 // tensor_dim0_stride lo
    g1[6] = 0;                                       // stride0 hi, stride1 lo
    g1[7] = 0;                                       // stride1 hi
    i32x4 z4 = {0, 0, 0, 0};
#if defined(TDM_6ARG)
    i32x8 z8 = {0, 0, 0, 0, 0, 0, 0, 0};
    __builtin_amdgcn_tensor_load_to_lds(g0, g1, z4, z4, z8, 0);
#else
    __builtin_amdgcn_tensor_load_to_lds(g0, g1, z4, z4, 0);
#endif
}
#endif

// Stage nbytes of weight fragments into LDS once per workgroup.
__device__ __forceinline__ void stage_weights(__bf16* smem, const __bf16* gsrc,
                                              unsigned nbytes)
{
#if defined(HAVE_TDM)
    if ((threadIdx.x >> 5) == 0) {                   // wave 0 issues one TDM op
        tdm_load_1d((unsigned)(uintptr_t)smem, gsrc, nbytes);
        __builtin_amdgcn_s_wait_tensorcnt(0);        // TENSORcnt is per-wave
    }
    __syncthreads();
#else
    unsigned long long* d = (unsigned long long*)smem;
    const unsigned long long* s = (const unsigned long long*)gsrc;
    const unsigned n64 = nbytes >> 3;
    for (unsigned i = threadIdx.x; i < n64; i += blockDim.x) d[i] = s[i];
    __syncthreads();
#endif
}

// ---------------------------------------------------------------------------
// Edge kernel: per wave, a 16-edge x 128-col tile of
//   relu(X @ Wmsg + b) + relu(X @ Wrev + b) , X = [nf[from] | nf[to] | ef]
// then scatter-add into agg rows (to_idx) with f32 atomics.
// Weights (192 KB of pre-swizzled bf16 fragments) are TDM-staged into LDS
// once per 16-wave block; B-fragments then come from the LDS banked path so
// the VMEM path is reserved for the random gathers + atomics.
// A 16x32 bf16 layout: lane l (m = l%16, hi = l/16) holds two runs of 8:
//   elements 0..7  -> k = hi*8 + 0..7 ; elements 8..15 -> k = 16 + hi*8 + 0..7
// ---------------------------------------------------------------------------
__global__ __launch_bounds__(512) void edge_msg_scatter(
    const float* __restrict__ nf, const float* __restrict__ ef,
    const int* __restrict__ from_idx, const int* __restrict__ to_idx,
    const __bf16* __restrict__ wfrag,   // wmsg frags, then wrev frags
    const float* __restrict__ bmsg, const float* __restrict__ brev,
    float* __restrict__ agg, int n_tiles)
{
    extern __shared__ __bf16 smem[];                 // 2*S_EDGE bf16 = 192 KB
    stage_weights(smem, wfrag, 2 * S_EDGE * (unsigned)sizeof(__bf16));

    const int lane = threadIdx.x & 31;
    const int wave = blockIdx.x * (blockDim.x >> 5) + (threadIdx.x >> 5);
    if (wave >= n_tiles) return;                     // wave-uniform, after barrier
    const int hi  = lane >> 4;
    const int row = lane & 15;
    const long e  = (long)wave * 16 + row;

    const long f = from_idx[e];
    const long t = to_idx[e];
    const float* base0 = nf + f * NODE_D;            // k in [0,128)
    const float* base1 = nf + t * NODE_D;            // k in [128,256)
    const float* base2 = ef + e * NODE_D;            // k in [256,384)

    v8f accm[8] = {};
    v8f accr[8] = {};

    #pragma unroll
    for (int ks = 0; ks < 12; ++ks) {
        const float* base = (ks < 4) ? base0 : ((ks < 8) ? base1 : base2);
        const float4* q = (const float4*)(base + (ks & 3) * 32 + hi * 8);
        float4 r0 = q[0], r1 = q[1];                 // k_local = hi*8 + 0..7
        float4 r2 = q[4], r3 = q[5];                 // k_local = 16 + hi*8 + 0..7
        v16bf a;
        a[0]=(__bf16)r0.x;  a[1]=(__bf16)r0.y;  a[2]=(__bf16)r0.z;  a[3]=(__bf16)r0.w;
        a[4]=(__bf16)r1.x;  a[5]=(__bf16)r1.y;  a[6]=(__bf16)r1.z;  a[7]=(__bf16)r1.w;
        a[8]=(__bf16)r2.x;  a[9]=(__bf16)r2.y;  a[10]=(__bf16)r2.z; a[11]=(__bf16)r2.w;
        a[12]=(__bf16)r3.x; a[13]=(__bf16)r3.y; a[14]=(__bf16)r3.z; a[15]=(__bf16)r3.w;

        const __bf16* wm = smem + (ks * 8) * 512 + lane * 16;
        const __bf16* wr = wm + S_EDGE;
        #pragma unroll
        for (int nt = 0; nt < 8; ++nt) {
            v16bf bm = *(const v16bf*)(wm + nt * 512);   // ds_load_b128 x2
            accm[nt] = __builtin_amdgcn_wmma_f32_16x16x32_bf16(
                false, a, false, bm, (short)0, accm[nt], false, false);
            v16bf br = *(const v16bf*)(wr + nt * 512);
            accr[nt] = __builtin_amdgcn_wmma_f32_16x16x32_bf16(
                false, a, false, br, (short)0, accr[nt], false, false);
        }
    }

    // Epilogue: D layout -> VGPR r: lane<16 row m=r, lane>=16 row m=8+r; col = lane%16.
    const int col0 = lane & 15;
    float bm8[8], br8[8];
    #pragma unroll
    for (int nt = 0; nt < 8; ++nt) {
        bm8[nt] = bmsg[nt * 16 + col0];
        br8[nt] = brev[nt * 16 + col0];
    }
    #pragma unroll
    for (int r = 0; r < 8; ++r) {
        const int m = hi * 8 + r;
        float* dst = agg + (long)to_idx[(long)wave * 16 + m] * NODE_D;
        #pragma unroll
        for (int nt = 0; nt < 8; ++nt) {
            float v = fmaxf(accm[nt][r] + bm8[nt], 0.f)
                    + fmaxf(accr[nt][r] + br8[nt], 0.f);
            __hip_atomic_fetch_add(dst + nt * 16 + col0, v,
                                   __ATOMIC_RELAXED, __HIP_MEMORY_SCOPE_AGENT);
        }
    }
}

// ---------------------------------------------------------------------------
// Node update: out = nf + relu(agg @ Wupd + b). agg aliases out (in-place):
// each wave reads its own 16 rows fully into registers before writing them.
// Wupd fragments (32 KB bf16) TDM-staged into LDS per block.
// ---------------------------------------------------------------------------
__global__ __launch_bounds__(512) void node_update(
    const float* __restrict__ nf, const __bf16* __restrict__ wfrag,
    const float* __restrict__ bupd, float* __restrict__ out, int n_tiles)
{
    extern __shared__ __bf16 smem[];                 // S_UPD bf16 = 32 KB
    stage_weights(smem, wfrag, S_UPD * (unsigned)sizeof(__bf16));

    const int lane = threadIdx.x & 31;
    const int wave = blockIdx.x * (blockDim.x >> 5) + (threadIdx.x >> 5);
    if (wave >= n_tiles) return;                     // wave-uniform, after barrier
    const int hi  = lane >> 4;
    const int row = lane & 15;
    const float* arow = out + ((long)wave * 16 + row) * NODE_D;  // agg (in-place)

    v16bf afrag[4];
    #pragma unroll
    for (int ks = 0; ks < 4; ++ks) {
        const float4* q = (const float4*)(arow + ks * 32 + hi * 8);
        float4 r0 = q[0], r1 = q[1], r2 = q[4], r3 = q[5];
        v16bf a;
        a[0]=(__bf16)r0.x;  a[1]=(__bf16)r0.y;  a[2]=(__bf16)r0.z;  a[3]=(__bf16)r0.w;
        a[4]=(__bf16)r1.x;  a[5]=(__bf16)r1.y;  a[6]=(__bf16)r1.z;  a[7]=(__bf16)r1.w;
        a[8]=(__bf16)r2.x;  a[9]=(__bf16)r2.y;  a[10]=(__bf16)r2.z; a[11]=(__bf16)r2.w;
        a[12]=(__bf16)r3.x; a[13]=(__bf16)r3.y; a[14]=(__bf16)r3.z; a[15]=(__bf16)r3.w;
        afrag[ks] = a;
    }

    v8f acc[8] = {};
    #pragma unroll
    for (int ks = 0; ks < 4; ++ks) {
        const __bf16* w = smem + (ks * 8) * 512 + lane * 16;
        #pragma unroll
        for (int nt = 0; nt < 8; ++nt) {
            v16bf b = *(const v16bf*)(w + nt * 512);
            acc[nt] = __builtin_amdgcn_wmma_f32_16x16x32_bf16(
                false, afrag[ks], false, b, (short)0, acc[nt], false, false);
        }
    }

    const int col0 = lane & 15;
    float b8[8];
    #pragma unroll
    for (int nt = 0; nt < 8; ++nt) b8[nt] = bupd[nt * 16 + col0];
    #pragma unroll
    for (int r = 0; r < 8; ++r) {
        const long mrow = (long)wave * 16 + hi * 8 + r;
        const float* nrow = nf + mrow * NODE_D;
        float* orow = out + mrow * NODE_D;
        #pragma unroll
        for (int nt = 0; nt < 8; ++nt) {
            const int col = nt * 16 + col0;
            orow[col] = nrow[col] + fmaxf(acc[nt][r] + b8[nt], 0.f);
        }
    }
}

extern "C" void kernel_launch(void* const* d_in, const int* in_sizes, int n_in,
                              void* d_out, int out_size, void* d_ws, size_t ws_size,
                              hipStream_t stream) {
    (void)n_in; (void)out_size; (void)ws_size;
    const float* nf   = (const float*)d_in[0];
    const float* ef   = (const float*)d_in[1];
    const int* fidx   = (const int*)d_in[2];
    const int* tidx   = (const int*)d_in[3];
    const float* Wmsg = (const float*)d_in[4];
    const float* bmsg = (const float*)d_in[5];
    const float* Wrev = (const float*)d_in[6];
    const float* brev = (const float*)d_in[7];
    const float* Wupd = (const float*)d_in[8];
    const float* bupd = (const float*)d_in[9];
    float* out  = (float*)d_out;
    __bf16* wsb = (__bf16*)d_ws;

    const int n_nodes = in_sizes[0] / NODE_D;    // 100000
    const int n_edges = in_sizes[2];             // 640000

    // agg accumulates in d_out (exactly n_nodes*128 f32); zero it each call.
    hipMemsetAsync(out, 0, (size_t)n_nodes * NODE_D * sizeof(float), stream);

    // pre-swizzle weights: 2*S_EDGE + S_UPD = 114688 elements
    prep_weights<<<(2 * S_EDGE + S_UPD + 255) / 256, 256, 0, stream>>>(Wmsg, Wrev, Wupd, wsb);

    const int etiles = n_edges / 16;             // 40000 waves, 16 waves/block
    edge_msg_scatter<<<(etiles + 15) / 16, 512, 2 * S_EDGE * sizeof(__bf16), stream>>>(
        nf, ef, fidx, tidx, wsb, bmsg, brev, out, etiles);

    const int ntiles = n_nodes / 16;             // 6250 waves
    node_update<<<(ntiles + 15) / 16, 512, S_UPD * sizeof(__bf16), stream>>>(
        nf, wsb + 2 * S_EDGE, bupd, out, ntiles);
}
